// AttnDecoderCell_67018669687087
// MI455X (gfx1250) — matrix operand order repaired and checked by reading, not compiled, not run on hardware
//
#include <hip/hip_runtime.h>
#include <hip/hip_bf16.h>
#include <stdint.h>

#define H 1024
#define L 4096
#define V 50257

typedef __attribute__((ext_vector_type(16))) __bf16 v16bf;
typedef __attribute__((ext_vector_type(8)))  float  v8f;

union FragBF { unsigned int u[8]; v16bf v; };

// ---------------- stage 0: precision-split conversions ----------------

__global__ void k_cvt_enc(const float* __restrict__ enc,
                          __bf16* __restrict__ eh, __bf16* __restrict__ el) {
    int i = blockIdx.x * 256 + threadIdx.x;       // L*H elements total
    float x = enc[i];
    __bf16 h = (__bf16)x;
    eh[i] = h;
    el[i] = (__bf16)(x - (float)h);
}

// WkT[j][h] = Wk[h][j], split into bf16 hi/lo.  32x32 LDS tile transpose.
__global__ void k_cvt_wkT(const float* __restrict__ Wk,
                          __bf16* __restrict__ th, __bf16* __restrict__ tl) {
    __shared__ float tile[32][33];
    int tx = threadIdx.x & 31, ty = threadIdx.x >> 5;   // ty 0..7
    int h0 = blockIdx.y * 32, j0 = blockIdx.x * 32;
#pragma unroll
    for (int yy = 0; yy < 4; ++yy) {
        int h = h0 + ty + yy * 8;
        tile[ty + yy * 8][tx] = Wk[h * H + j0 + tx];
    }
    __syncthreads();
#pragma unroll
    for (int yy = 0; yy < 4; ++yy) {
        int j = j0 + ty + yy * 8;
        float x = tile[tx][ty + yy * 8];
        __bf16 hh = (__bf16)x;
        th[j * H + h0 + tx] = hh;
        tl[j * H + h0 + tx] = (__bf16)(x - (float)hh);
    }
}

// embedding relu + qWq matvec + pack (qWq[j], wv[j]) for the GEMM epilogue
__global__ void k_embed_qwq(const int* __restrict__ tok,
                            const float* __restrict__ emb,
                            const float* __restrict__ hidden,
                            const float* __restrict__ Wq,
                            const float* __restrict__ wv,
                            float* __restrict__ embv,
                            float2* __restrict__ qwv) {
    __shared__ float qs[H];
    int tid = threadIdx.x;
    for (int i = tid; i < H; i += 256) qs[i] = hidden[H + i];   // last layer
    __syncthreads();
    int j = blockIdx.x * 256 + tid;
    long t = (long)tok[0];
    embv[j] = fmaxf(emb[t * H + j], 0.f);
    float acc = 0.f;
    for (int h = 0; h < H; ++h) acc = fmaf(qs[h], Wq[h * H + j], acc);
    qwv[j] = make_float2(acc, wv[j]);
}

// ---------------- stage 1: attention GEMM via split-bf16 WMMA ----------------
// grid (L/128, H/128); block 256 = 8 waves; wave w owns rows [w*16, w*16+16)
// of the 128-row tile, all 128 columns (8 col tiles).  Full K=1024 per block.
// Tiles staged with double-buffered GLOBAL_LOAD_ASYNC_TO_LDS_B128 (ASYNCcnt):
// next k-step's tiles stream into the alternate LDS buffer while the current
// step's 24 WMMAs run.  Emits per-row partial scores -> scorePart[cb][l].
__global__ __launch_bounds__(256) void k_attn_gemm(
        const __bf16* __restrict__ eh,  const __bf16* __restrict__ el,
        const __bf16* __restrict__ wth, const __bf16* __restrict__ wtl,
        const float2* __restrict__ qwv, float* __restrict__ scorePart) {
    __shared__ unsigned int sAh[2][2048], sAl[2][2048];     // 64 KB total
    __shared__ unsigned int sBh[2][2048], sBl[2][2048];

    int tid  = threadIdx.x;
    int lane = tid & 31, wq = tid >> 5;
    int m = lane & 15, g = lane >> 4;
    int l0 = blockIdx.x * 128;
    int cb = blockIdx.y;
    int c0 = cb * 128;

    // issue one k-step's tiles (128x32 bf16 hi+lo of A and B) as 8 async
    // 16-byte memory->LDS transfers per thread
    auto issue_tiles = [&](int ks, int buf) {
#pragma unroll
        for (int i = 0; i < 2; ++i) {
            int c   = tid + i * 256;            // 16B chunk id, 0..511
            int row = c >> 2, q = c & 3;
            unsigned gA = (unsigned)((((l0 + row) * H + ks * 32) << 1) + (q << 4));
            unsigned gB = (unsigned)((((c0 + row) * H + ks * 32) << 1) + (q << 4));
            unsigned lAh = (unsigned)(uintptr_t)&sAh[buf][(row << 4) + (q << 2)];
            unsigned lAl = (unsigned)(uintptr_t)&sAl[buf][(row << 4) + (q << 2)];
            unsigned lBh = (unsigned)(uintptr_t)&sBh[buf][(row << 4) + (q << 2)];
            unsigned lBl = (unsigned)(uintptr_t)&sBl[buf][(row << 4) + (q << 2)];
            asm volatile("global_load_async_to_lds_b128 %0, %1, %2"
                         :: "v"(lAh), "v"(gA), "s"(eh)  : "memory");
            asm volatile("global_load_async_to_lds_b128 %0, %1, %2"
                         :: "v"(lAl), "v"(gA), "s"(el)  : "memory");
            asm volatile("global_load_async_to_lds_b128 %0, %1, %2"
                         :: "v"(lBh), "v"(gB), "s"(wth) : "memory");
            asm volatile("global_load_async_to_lds_b128 %0, %1, %2"
                         :: "v"(lBl), "v"(gB), "s"(wtl) : "memory");
        }
    };

    v8f acc[8];
#pragma unroll
    for (int t = 0; t < 8; ++t)
#pragma unroll
        for (int e = 0; e < 8; ++e) acc[t][e] = 0.f;

    issue_tiles(0, 0);                       // prime the pipeline

    for (int ks = 0; ks < 32; ++ks) {
        int buf = ks & 1;
        if (ks < 31) {
            issue_tiles(ks + 1, buf ^ 1);    // prefetch next step
            // async loads complete in order: <=8 outstanding means the
            // current buffer's batch (issued last iteration) has landed
            asm volatile("s_wait_asynccnt 0x8" ::: "memory");
        } else {
            asm volatile("s_wait_asynccnt 0x0" ::: "memory");
        }
        __syncthreads();

        // A fragment: ISA layout, lanes0-15: K 0-7 (v0-3), 16-23 (v4-7);
        //             lanes16-31: K 8-15, 24-31
        FragBF ah, al;
        int arow = wq * 16 + m;
#pragma unroll
        for (int i = 0; i < 8; ++i) {
            int kb = ((i >> 2) << 4) + (g << 3) + ((i & 3) << 1);
            ah.u[i] = sAh[buf][arow * 16 + (kb >> 1)];
            al.u[i] = sAl[buf][arow * 16 + (kb >> 1)];
        }
#pragma unroll
        for (int t = 0; t < 8; ++t) {
            // B fragment: same layout as A (lanes0-15: K 0-7, 16-23; lanes16-31: K 8-15, 24-31)
            FragBF bh, bl;
            int brow = t * 16 + m;
#pragma unroll
            for (int i = 0; i < 8; ++i) {
                int kb = ((i >> 2) << 4) + (g << 3) + ((i & 3) << 1);
                bh.u[i] = sBh[buf][brow * 16 + (kb >> 1)];
                bl.u[i] = sBl[buf][brow * 16 + (kb >> 1)];
            }
            // split-precision: hi*hi + hi*lo + lo*hi  (~f32 accuracy)
            acc[t] = __builtin_amdgcn_wmma_f32_16x16x32_bf16(
                        false, ah.v, false, bh.v, (short)0, acc[t], false, false);
            acc[t] = __builtin_amdgcn_wmma_f32_16x16x32_bf16(
                        false, ah.v, false, bl.v, (short)0, acc[t], false, false);
            acc[t] = __builtin_amdgcn_wmma_f32_16x16x32_bf16(
                        false, al.v, false, bh.v, (short)0, acc[t], false, false);
        }
        __syncthreads();   // all readers done before this buffer is re-filled
    }

    // epilogue: s_lj = tanh(c + qWq[j]) * wv[j]; reduce over j (this cb slice)
    float rp[8];
#pragma unroll
    for (int v = 0; v < 8; ++v) rp[v] = 0.f;
#pragma unroll
    for (int t = 0; t < 8; ++t) {
        float2 qw = qwv[c0 + t * 16 + m];       // column = N = m
#pragma unroll
        for (int v = 0; v < 8; ++v)
            rp[v] += tanhf(acc[t][v] + qw.x) * qw.y;
    }
#pragma unroll
    for (int v = 0; v < 8; ++v) {
        float s = rp[v];
#pragma unroll
        for (int off = 8; off > 0; off >>= 1) s += __shfl_xor(s, off, 32);
        if (m == 0)   // row M = v + 8*g within wave tile
            scorePart[cb * L + l0 + wq * 16 + g * 8 + v] = s;
    }
}

// ---------------- stage 2: softmax over L (single block, deterministic) ----
__global__ void k_softmax(const float* __restrict__ sp, float* __restrict__ attn) {
    __shared__ float rb[1024];
    int tid = threadIdx.x;
    float s[4]; float mx = -3.4e38f;
#pragma unroll
    for (int i = 0; i < 4; ++i) {
        int l = tid + i * 1024;
        float a = 0.f;
#pragma unroll
        for (int cb = 0; cb < 8; ++cb) a += sp[cb * L + l];
        s[i] = a; mx = fmaxf(mx, a);
    }
    rb[tid] = mx; __syncthreads();
    for (int o = 512; o > 0; o >>= 1) { if (tid < o) rb[tid] = fmaxf(rb[tid], rb[tid + o]); __syncthreads(); }
    mx = rb[0]; __syncthreads();
    float sum = 0.f;
#pragma unroll
    for (int i = 0; i < 4; ++i) { s[i] = expf(s[i] - mx); sum += s[i]; }
    rb[tid] = sum; __syncthreads();
    for (int o = 512; o > 0; o >>= 1) { if (tid < o) rb[tid] += rb[tid + o]; __syncthreads(); }
    float inv = 1.f / rb[0];
#pragma unroll
    for (int i = 0; i < 4; ++i) attn[tid + i * 1024] = s[i] * inv;
}

// ---------------- stage 3: context = attn^T enc (two-stage, no atomics) ----
__global__ void k_ctx_part(const float* __restrict__ attn,
                           const float* __restrict__ enc,
                           float* __restrict__ part) {
    int j = blockIdx.x * 256 + threadIdx.x;
    int cy = blockIdx.y, lb = cy * 256;
    float acc = 0.f;
    for (int l = 0; l < 256; ++l) acc = fmaf(attn[lb + l], enc[(lb + l) * H + j], acc);
    part[cy * H + j] = acc;
}

__global__ void k_ctx_final(const float* __restrict__ part,
                            const float* __restrict__ embv,
                            float* __restrict__ x) {
    int j = blockIdx.x * 256 + threadIdx.x;
    float acc = 0.f;
#pragma unroll
    for (int c = 0; c < 16; ++c) acc += part[c * H + j];
    x[j] = acc;          // x = concat(context, emb_vec)
    x[H + j] = embv[j];
}

// ---------------- stage 4: GRU mat-vecs (wave-per-row, coalesced) ----------
__global__ __launch_bounds__(256) void k_matvec(
        const float* __restrict__ W, const float* __restrict__ b,
        const float* __restrict__ x, float* __restrict__ y, int cols) {
    __shared__ float xs[2048];
    int tid = threadIdx.x;
    for (int c = tid; c < cols; c += 256) xs[c] = x[c];
    __syncthreads();
    int lane = tid & 31, wq = tid >> 5;
    int r = blockIdx.x * 8 + wq;
    const float* Wr = W + (long)r * cols;
    float acc = 0.f;
    for (int c = lane; c < cols; c += 32) acc = fmaf(Wr[c], xs[c], acc);
#pragma unroll
    for (int off = 16; off > 0; off >>= 1) acc += __shfl_xor(acc, off, 32);
    if (lane == 0) y[r] = acc + b[r];
}

__global__ void k_gru_combine(const float* __restrict__ gi, const float* __restrict__ gh,
                              const float* __restrict__ hp,
                              float* __restrict__ hn_ws, float* __restrict__ hn_out) {
    int j = blockIdx.x * 256 + threadIdx.x;
    float r = 1.f / (1.f + expf(-(gi[j] + gh[j])));
    float z = 1.f / (1.f + expf(-(gi[H + j] + gh[H + j])));
    float n = tanhf(gi[2 * H + j] + r * gh[2 * H + j]);
    float h = (1.f - z) * n + z * hp[j];
    hn_ws[j] = h; hn_out[j] = h;
}

// ---------------- stage 5: vocab projection + log_softmax ------------------
// Wd (206 MB) is streamed exactly once and exceeds the 192 MB L2 ->
// non-temporal loads keep it from thrashing the cache.
__global__ __launch_bounds__(256) void k_logits(
        const float* __restrict__ h1, const float* __restrict__ Wd,
        const float* __restrict__ bd, float* __restrict__ out) {
    __shared__ float hs[H];
    int tid = threadIdx.x;
#pragma unroll
    for (int i = 0; i < 4; ++i) hs[tid + i * 256] = h1[tid + i * 256];
    __syncthreads();
    int v = blockIdx.x * 256 + tid;
    if (v >= V) return;
    float acc = bd[v];
    const float* wp = Wd + v;
#pragma unroll 4
    for (int h = 0; h < H; ++h)
        acc = fmaf(hs[h], __builtin_nontemporal_load(wp + (long)h * V), acc);
    out[v] = acc;
}

__global__ void k_logred(const float* __restrict__ logit, float* __restrict__ red) {
    __shared__ float rb[1024];
    int tid = threadIdx.x;
    float mx = -3.4e38f;
    for (int v = tid; v < V; v += 1024) mx = fmaxf(mx, logit[v]);
    rb[tid] = mx; __syncthreads();
    for (int o = 512; o > 0; o >>= 1) { if (tid < o) rb[tid] = fmaxf(rb[tid], rb[tid + o]); __syncthreads(); }
    mx = rb[0]; __syncthreads();
    float sum = 0.f;
    for (int v = tid; v < V; v += 1024) sum += expf(logit[v] - mx);
    rb[tid] = sum; __syncthreads();
    for (int o = 512; o > 0; o >>= 1) { if (tid < o) rb[tid] += rb[tid + o]; __syncthreads(); }
    if (tid == 0) red[0] = mx + logf(rb[0]);
}

__global__ void k_logapply(float* __restrict__ out, const float* __restrict__ red) {
    int v = blockIdx.x * 256 + threadIdx.x;
    if (v < V) out[v] -= red[0];
}

// ---------------- host: launch sequence ------------------------------------
extern "C" void kernel_launch(void* const* d_in, const int* in_sizes, int n_in,
                              void* d_out, int out_size, void* d_ws, size_t ws_size,
                              hipStream_t stream) {
    const int*   tok    = (const int*)  d_in[0];
    const float* hidden = (const float*)d_in[1];   // [2,1,H]
    const float* enc    = (const float*)d_in[2];   // [L,H]
    const float* emb    = (const float*)d_in[3];   // [V,H]
    const float* Wq     = (const float*)d_in[4];
    const float* Wk     = (const float*)d_in[5];
    const float* wv     = (const float*)d_in[6];
    const float* W_ih0  = (const float*)d_in[7];
    const float* W_hh0  = (const float*)d_in[8];
    const float* b_ih0  = (const float*)d_in[9];
    const float* b_hh0  = (const float*)d_in[10];
    const float* W_ih1  = (const float*)d_in[11];
    const float* W_hh1  = (const float*)d_in[12];
    const float* b_ih1  = (const float*)d_in[13];
    const float* b_hh1  = (const float*)d_in[14];
    const float* Wd     = (const float*)d_in[15];
    const float* bd     = (const float*)d_in[16];
    float* out = (float*)d_out;                    // [V] logp, then [2H] hidden

    char* ws = (char*)d_ws;
    size_t o = 0;
    auto alloc = [&](size_t bytes) -> void* {
        void* p = ws + o;
        o = (o + bytes + 255) & ~((size_t)255);
        return p;
    };
    __bf16* encH = (__bf16*)alloc((size_t)L * H * 2);
    __bf16* encL = (__bf16*)alloc((size_t)L * H * 2);
    __bf16* WkTH = (__bf16*)alloc((size_t)H * H * 2);
    __bf16* WkTL = (__bf16*)alloc((size_t)H * H * 2);
    float2* qwv  = (float2*)alloc((size_t)H * 8);
    float*  embv = (float*) alloc((size_t)H * 4);
    float*  sPart= (float*) alloc((size_t)8 * L * 4);
    float*  attn = (float*) alloc((size_t)L * 4);
    float*  cPart= (float*) alloc((size_t)16 * H * 4);
    float*  xvec = (float*) alloc((size_t)2 * H * 4);
    float*  gi0  = (float*) alloc((size_t)3 * H * 4);
    float*  gh0  = (float*) alloc((size_t)3 * H * 4);
    float*  gi1  = (float*) alloc((size_t)3 * H * 4);
    float*  gh1  = (float*) alloc((size_t)3 * H * 4);
    float*  h0n  = (float*) alloc((size_t)H * 4);
    float*  h1n  = (float*) alloc((size_t)H * 4);
    float*  red  = (float*) alloc(256);

    k_cvt_enc  <<<(L * H) / 256, 256, 0, stream>>>(enc, encH, encL);
    k_cvt_wkT  <<<dim3(32, 32), 256, 0, stream>>>(Wk, WkTH, WkTL);
    k_embed_qwq<<<H / 256, 256, 0, stream>>>(tok, emb, hidden, Wq, wv, embv, qwv);

    k_attn_gemm<<<dim3(L / 128, 8), 256, 0, stream>>>(encH, encL, WkTH, WkTL, qwv, sPart);
    k_softmax  <<<1, 1024, 0, stream>>>(sPart, attn);
    k_ctx_part <<<dim3(H / 256, 16), 256, 0, stream>>>(attn, enc, cPart);
    k_ctx_final<<<H / 256, 256, 0, stream>>>(cPart, embv, xvec);

    k_matvec     <<<384, 256, 0, stream>>>(W_ih0, b_ih0, xvec, gi0, 2 * H);
    k_matvec     <<<384, 256, 0, stream>>>(W_hh0, b_hh0, hidden, gh0, H);
    k_gru_combine<<<H / 256, 256, 0, stream>>>(gi0, gh0, hidden, h0n, out + V);
    k_matvec     <<<384, 256, 0, stream>>>(W_ih1, b_ih1, h0n, gi1, H);
    k_matvec     <<<384, 256, 0, stream>>>(W_hh1, b_hh1, hidden + H, gh1, H);
    k_gru_combine<<<H / 256, 256, 0, stream>>>(gi1, gh1, hidden + H, h1n, out + V + H);

    k_logits  <<<(V + 255) / 256, 256, 0, stream>>>(h1n, Wd, bd, out);
    k_logred  <<<1, 1024, 0, stream>>>(out, red);
    k_logapply<<<(V + 255) / 256, 256, 0, stream>>>(out, red);
}
